// DeepKoopmanLinDec_48112223650187
// MI455X (gfx1250) — matrix operand order repaired and checked by reading, not compiled, not run on hardware
//
#include <hip/hip_runtime.h>

// ---------------------------------------------------------------------------
// DeepKoopman (linear decoder) for MI455X / gfx1250, wave32 + WMMA f16.
// B=2048, M=64, SD=32, CD=8, LD=128, HW=512
// Compute-bound on the encoder (~161 GFLOP of the ~167 total) => WMMA f16
// with f32 accumulate; A-fragment register reuse across 4 N-blocks.
// ---------------------------------------------------------------------------

#define B_N   2048
#define M_N   64
#define SD_N  32
#define CD_N  8
#define LD_N  128
#define HW_N  512

typedef __attribute__((ext_vector_type(16))) _Float16 v16h;
typedef __attribute__((ext_vector_type(8)))  _Float16 v8h;
typedef __attribute__((ext_vector_type(8)))  float    v8f;

// ---- WMMA wrapper: D = A(16x32 f16) * B(32x16 f16) + C(16x16 f32) ----------
static __device__ __forceinline__ v8f wmma_f16(v16h a, v16h b, v8f c) {
  return __builtin_amdgcn_wmma_f32_16x16x32_f16(
      /*neg_a=*/false, a, /*neg_b=*/false, b,
      /*c_mod=*/(short)0, c, /*reuse_a=*/false, /*reuse_b=*/false);
}

// ---- A-fragment (16x32) from f16 row-major buffer (stride ld halfs) --------
// ISA layout: lane row = lane%16; half h -> K = (h<8 ? h : h+8) + 8*(lane/16)
static __device__ __forceinline__ v16h load_a_f16(const _Float16* hsrc,
                                                  int row0, int ld, int k0) {
  int lane = threadIdx.x & 31;
  int hi   = lane >> 4;
  const _Float16* p = hsrc + (size_t)(row0 + (lane & 15)) * ld + k0 + 8 * hi;
  v8h c0 = *(const v8h*)(p);        // K = k0 + 8*hi + 0..7
  v8h c1 = *(const v8h*)(p + 16);   // K = k0 + 16 + 8*hi + 0..7
  v16h a;
#pragma unroll
  for (int i = 0; i < 8; ++i) { a[i] = c0[i]; a[8 + i] = c1[i]; }
  return a;
}

// ---- A-fragment (16x32) from f32 row-major buffer (LDS or global) ----------
static __device__ __forceinline__ v16h load_a_f32(const float* fsrc,
                                                  int row0, int ld, int k0) {
  int lane = threadIdx.x & 31;
  int hi   = lane >> 4;
  const float* p = fsrc + (size_t)(row0 + (lane & 15)) * ld + k0 + 8 * hi;
  v8f c0 = *(const v8f*)(p);
  v8f c1 = *(const v8f*)(p + 16);
  v16h a;
#pragma unroll
  for (int i = 0; i < 8; ++i) {
    a[i]     = (_Float16)c0[i];
    a[8 + i] = (_Float16)c1[i];
  }
  return a;
}

// ---- B-fragment (32x16) from pre-transposed weights Wt[N,K] (stride ldk) ---
// ISA layout: N = lane%16, half h -> K = k0 + (h<8 ? h : h+8) + 8*(lane/16)
static __device__ __forceinline__ v16h load_b_wt(const _Float16* wt,
                                                 int n0, int ldk, int k0) {
  int lane = threadIdx.x & 31;
  const _Float16* p = wt + (size_t)(n0 + (lane & 15)) * ldk + k0 + 8 * (lane >> 4);
  v8h c0 = *(const v8h*)(p);
  v8h c1 = *(const v8h*)(p + 16);
  v16h b;
#pragma unroll
  for (int i = 0; i < 8; ++i) { b[i] = c0[i]; b[8 + i] = c1[i]; }
  return b;
}

// ---- D-store (16x16 f32) to f16 LDS with bias + ReLU -----------------------
// ISA layout: N = lane%16, VGPR v -> M = v + 8*(lane/16)
static __device__ __forceinline__ void store_d_lds_relu(v8f acc, _Float16* hdst,
                                                        int row0, int ld, int n0,
                                                        const float* __restrict__ bias) {
  int lane = threadIdx.x & 31;
  int hi   = lane >> 4;
  int n    = n0 + (lane & 15);
  float bv = bias[n];
#pragma unroll
  for (int v = 0; v < 8; ++v) {
    float x = acc[v] + bv;
    x = x > 0.f ? x : 0.f;
    hdst[(size_t)(row0 + v + 8 * hi) * ld + n] = (_Float16)x;
  }
}

// ---- D-store (16x16 f32) to global f32 with bias (no ReLU) -----------------
static __device__ __forceinline__ void store_d_global(v8f acc, float* __restrict__ out,
                                                      long row0, int ld, int n0,
                                                      const float* __restrict__ bias) {
  int lane = threadIdx.x & 31;
  int hi   = lane >> 4;
  int n    = n0 + (lane & 15);
  float bv = bias[n];
#pragma unroll
  for (int v = 0; v < 8; ++v)
    out[(size_t)(row0 + v + 8 * hi) * ld + n] = acc[v] + bv;
}

// ---------------------------------------------------------------------------
// Kernel: transpose + convert f32 weight [K,N] -> f16 Wt [N,K]
// ---------------------------------------------------------------------------
__global__ void tcvt_kernel(const float* __restrict__ src, _Float16* __restrict__ dst,
                            int K, int N) {
  int i = blockIdx.x * 256 + threadIdx.x;
  if (i < K * N) {
    int k = i / N, n = i % N;
    dst[(size_t)n * K + k] = (_Float16)src[i];
  }
}

// ---------------------------------------------------------------------------
// One MLP layer on a 64-row tile. Each wave owns one M-block and sweeps
// groups of 4 N-blocks, so one A-fragment feeds 4 WMMAs (register reuse).
//   KDIM      : reduction depth (32 or 512)
//   NOUT      : output width (128 or 512)
//   TO_GLOBAL : write f32 to global (layer 3) instead of f16+ReLU to LDS
// ---------------------------------------------------------------------------
template <int KDIM, int NOUT, bool TO_GLOBAL>
static __device__ __forceinline__ void mlp_layer(
    const _Float16* hin, int ldin,
    const _Float16* __restrict__ wt, const float* __restrict__ bias,
    _Float16* hout, float* __restrict__ gout, long grow0, int gld, int wave) {
  const int mb = (wave & 3) * 16;
  for (int g = (wave >> 2); g < NOUT / 64; g += 2) {
    const int nb0 = g * 64;
    v8f acc[4] = {{}, {}, {}, {}};
#pragma unroll
    for (int k = 0; k < KDIM; k += 32) {
      v16h a = load_a_f16(hin, mb, ldin, k);      // shared by 4 WMMAs
#pragma unroll
      for (int j = 0; j < 4; ++j) {
        v16h b = load_b_wt(wt, nb0 + j * 16, KDIM, k);
        acc[j] = wmma_f16(a, b, acc[j]);
      }
    }
#pragma unroll
    for (int j = 0; j < 4; ++j) {
      if (TO_GLOBAL)
        store_d_global(acc[j], gout, grow0 + mb, gld, nb0 + j * 16, bias);
      else
        store_d_lds_relu(acc[j], hout, mb, NOUT, nb0 + j * 16, bias);
    }
  }
}

// ---------------------------------------------------------------------------
// Kernel: fused 4-layer encoder. Grid = nrows/64 blocks of 256 threads.
// x[nrows,32] -> z[nrows,128]
// ---------------------------------------------------------------------------
__global__ __launch_bounds__(256) void encoder_kernel(
    const float* __restrict__ x,
    const _Float16* __restrict__ w0t, const _Float16* __restrict__ w1t,
    const _Float16* __restrict__ w2t, const _Float16* __restrict__ w3t,
    const float* __restrict__ b0, const float* __restrict__ b1,
    const float* __restrict__ b2, const float* __restrict__ b3,
    float* __restrict__ zout) {
  __shared__ _Float16 xs[64 * SD_N];   //   4 KB
  __shared__ _Float16 hA[64 * HW_N];   //  64 KB
  __shared__ _Float16 hB[64 * HW_N];   //  64 KB
  const int row0 = blockIdx.x * 64;
  const int tid  = threadIdx.x;
  const int wave = tid >> 5;

  // warm a slice of the L2-resident weights (global_prefetch_b8, counter-free)
  __builtin_prefetch(w1t + (size_t)tid * 64, 0, 1);
  __builtin_prefetch(w2t + (size_t)tid * 64, 0, 1);
  __builtin_prefetch(w3t + (size_t)tid * 64, 0, 1);

  // stage x tile -> f16 LDS
  for (int i = tid; i < 64 * SD_N; i += 256)
    xs[i] = (_Float16)x[(size_t)row0 * SD_N + i];
  __syncthreads();

  mlp_layer<SD_N, HW_N, false>(xs, SD_N, w0t, b0, hA, nullptr, 0, 0, wave);
  __syncthreads();
  mlp_layer<HW_N, HW_N, false>(hA, HW_N, w1t, b1, hB, nullptr, 0, 0, wave);
  __syncthreads();
  mlp_layer<HW_N, HW_N, false>(hB, HW_N, w2t, b2, hA, nullptr, 0, 0, wave);
  __syncthreads();
  mlp_layer<HW_N, LD_N, true>(hA, HW_N, w3t, b3, nullptr, zout, (long)row0, LD_N, wave);
}

// ---------------------------------------------------------------------------
// Kernel: linear recurrence z_{m+1} = z_m @ A + u_m @ B, 64 steps.
// Rows are independent: each workgroup owns 64 rows, A^T fragments live in
// registers, z double-buffered in LDS (f32). Writes z_pred [B,M,LD].
// ---------------------------------------------------------------------------
__global__ __launch_bounds__(256) void scan_kernel(
    const float* __restrict__ zk,       // [B, LD]
    const float* __restrict__ u,        // [B, M, CD]
    const _Float16* __restrict__ At,    // A^T [LD, LD]
    const _Float16* __restrict__ Bwt,   // B^T [LD, CD]
    float* __restrict__ zpred) {        // [B, M, LD]
  __shared__ float zA[64 * LD_N];       // 32 KB
  __shared__ float zB[64 * LD_N];       // 32 KB
  const int b0row = blockIdx.x * 64;
  const int tid   = threadIdx.x;
  const int wave  = tid >> 5;           // wave w owns N-block w*16
  const int lane  = tid & 31;
  const int hi    = lane >> 4;
  const int ncol  = wave * 16 + (lane & 15);   // global output column 0..127

  // stage z_k tile (contiguous rows)
  for (int i = tid; i < 64 * LD_N; i += 256)
    zA[i] = zk[(size_t)b0row * LD_N + i];

  // preload A^T B-fragments for this wave's N-block (registers, reused 64x)
  v16h bf[4];
#pragma unroll
  for (int k = 0; k < 4; ++k) bf[k] = load_b_wt(At, wave * 16, LD_N, k * 32);

  // per-lane B_w column (K=8) as f32
  v8h bwh = *(const v8h*)(Bwt + (size_t)ncol * CD_N);
  float bwf[8];
#pragma unroll
  for (int c = 0; c < 8; ++c) bwf[c] = (float)bwh[c];

  __syncthreads();

  float* zcur = zA;
  float* znxt = zB;
  for (int m = 0; m < M_N; ++m) {
#pragma unroll
    for (int mb = 0; mb < 4; ++mb) {
      v8f acc;
      // seed accumulator with Bu = u[r,m,:] . B_w[:,ncol]  (K=8, VALU)
#pragma unroll
      for (int v = 0; v < 8; ++v) {
        int r = b0row + mb * 16 + v + 8 * hi;
        v8f uv = *(const v8f*)(u + ((size_t)r * M_N + m) * CD_N);
        float s = 0.f;
#pragma unroll
        for (int c = 0; c < 8; ++c) s += uv[c] * bwf[c];
        acc[v] = s;
      }
      // z @ A : 4 WMMAs over K=128
#pragma unroll
      for (int k = 0; k < 4; ++k) {
        v16h a = load_a_f32(zcur, mb * 16, LD_N, k * 32);
        acc = wmma_f16(a, bf[k], acc);
      }
      // write next-z LDS buffer (f32) + z_pred global
#pragma unroll
      for (int v = 0; v < 8; ++v) {
        int rr = mb * 16 + v + 8 * hi;
        float val = acc[v];
        znxt[(size_t)rr * LD_N + ncol] = val;
        zpred[((size_t)(b0row + rr) * M_N + m) * LD_N + ncol] = val;
      }
    }
    __syncthreads();
    float* t = zcur; zcur = znxt; znxt = t;
  }
}

// ---------------------------------------------------------------------------
// Kernel: decoder x_pred = z_pred @ C_w + C_b. [131072,128]@[128,32].
// 128-row workgroups; wave owns one M-block and BOTH N-blocks, so each
// A-fragment feeds 2 WMMAs. Grid = (B*M)/128.
// ---------------------------------------------------------------------------
__global__ __launch_bounds__(256) void decoder_kernel(
    const float* __restrict__ zpred,    // [B*M, LD]
    const _Float16* __restrict__ Cwt,   // C^T [SD, LD]
    const float* __restrict__ Cb,
    float* __restrict__ xpred) {        // [B*M, SD]
  const long row0 = (long)blockIdx.x * 128;
  const int wave = threadIdx.x >> 5;
  const int mb = wave * 16;             // 8 waves cover 128 rows
  v8f acc[2] = {{}, {}};
#pragma unroll
  for (int k = 0; k < LD_N; k += 32) {
    v16h a = load_a_f32(zpred + row0 * LD_N, mb, LD_N, k);
#pragma unroll
    for (int j = 0; j < 2; ++j) {
      v16h b = load_b_wt(Cwt, j * 16, LD_N, k);
      acc[j] = wmma_f16(a, b, acc[j]);
    }
  }
#pragma unroll
  for (int j = 0; j < 2; ++j)
    store_d_global(acc[j], xpred, row0 + mb, SD_N, j * 16, Cb);
}

// ---------------------------------------------------------------------------
// Launch
// ---------------------------------------------------------------------------
extern "C" void kernel_launch(void* const* d_in, const int* in_sizes, int n_in,
                              void* d_out, int out_size, void* d_ws, size_t ws_size,
                              hipStream_t stream) {
  (void)in_sizes; (void)n_in; (void)out_size; (void)ws_size;

  const float* x_k    = (const float*)d_in[0];
  const float* u_seq  = (const float*)d_in[1];
  const float* x_next = (const float*)d_in[2];
  const float* enc_w0 = (const float*)d_in[3];
  const float* enc_b0 = (const float*)d_in[4];
  const float* enc_w1 = (const float*)d_in[5];
  const float* enc_b1 = (const float*)d_in[6];
  const float* enc_w2 = (const float*)d_in[7];
  const float* enc_b2 = (const float*)d_in[8];
  const float* enc_w3 = (const float*)d_in[9];
  const float* enc_b3 = (const float*)d_in[10];
  const float* A_w    = (const float*)d_in[11];
  const float* B_w    = (const float*)d_in[12];
  const float* C_w    = (const float*)d_in[13];
  const float* C_b    = (const float*)d_in[14];

  // workspace layout (bytes)
  char* ws = (char*)d_ws;
  _Float16* w0t = (_Float16*)(ws + 0);                // 512x32   ->  32768 B
  _Float16* w1t = (_Float16*)(ws + 32768);            // 512x512  -> 524288 B
  _Float16* w2t = (_Float16*)(ws + 557056);           // 512x512  -> 524288 B
  _Float16* w3t = (_Float16*)(ws + 1081344);          // 128x512  -> 131072 B
  _Float16* At  = (_Float16*)(ws + 1212416);          // 128x128  ->  32768 B
  _Float16* Bwt = (_Float16*)(ws + 1245184);          // 128x8    ->   2048 B
  _Float16* Cwt = (_Float16*)(ws + 1247232);          // 32x128   ->   8192 B
  float*    zk  = (float*)(ws + 1255424);             // 2048x128 f32

  // output layout: z_pred [B,M,LD] | x_pred [B,M,SD] | z_target [B,M,LD]
  float* zpred = (float*)d_out;
  float* xpred = zpred + (size_t)B_N * M_N * LD_N;
  float* ztarg = xpred + (size_t)B_N * M_N * SD_N;

  // 1) weight transpose + f32->f16 convert
  tcvt_kernel<<<(SD_N * HW_N + 255) / 256, 256, 0, stream>>>(enc_w0, w0t, SD_N, HW_N);
  tcvt_kernel<<<(HW_N * HW_N + 255) / 256, 256, 0, stream>>>(enc_w1, w1t, HW_N, HW_N);
  tcvt_kernel<<<(HW_N * HW_N + 255) / 256, 256, 0, stream>>>(enc_w2, w2t, HW_N, HW_N);
  tcvt_kernel<<<(HW_N * LD_N + 255) / 256, 256, 0, stream>>>(enc_w3, w3t, HW_N, LD_N);
  tcvt_kernel<<<(LD_N * LD_N + 255) / 256, 256, 0, stream>>>(A_w, At, LD_N, LD_N);
  tcvt_kernel<<<(CD_N * LD_N + 255) / 256, 256, 0, stream>>>(B_w, Bwt, CD_N, LD_N);
  tcvt_kernel<<<(LD_N * SD_N + 255) / 256, 256, 0, stream>>>(C_w, Cwt, LD_N, SD_N);

  // 2) encoder on x_k -> z_k (scratch) and on x_next_seq -> z_target (d_out)
  encoder_kernel<<<B_N / 64, 256, 0, stream>>>(
      x_k, w0t, w1t, w2t, w3t, enc_b0, enc_b1, enc_b2, enc_b3, zk);
  encoder_kernel<<<(B_N * M_N) / 64, 256, 0, stream>>>(
      x_next, w0t, w1t, w2t, w3t, enc_b0, enc_b1, enc_b2, enc_b3, ztarg);

  // 3) linear recurrence -> z_pred
  scan_kernel<<<B_N / 64, 256, 0, stream>>>(zk, u_seq, At, Bwt, zpred);

  // 4) linear decoder -> x_pred
  decoder_kernel<<<(B_N * M_N) / 128, 256, 0, stream>>>(zpred, Cwt, C_b, xpred);
}
